// GatedMultiLoRALayer_23888608101092
// MI455X (gfx1250) — compile-verified, matrix operand order checked by reading, not run on hardware
//
#include <hip/hip_runtime.h>
#include <hip/hip_bf16.h>

// ---------------------------------------------------------------------------
// GatedMultiLoRA fused layer for MI455X (gfx1250, wave32, WMMA + TDM).
//
// out = softmax-gated mix of base GEMM + 8 LoRA experts. Folded into:
//   out = (g0 (.) x) @ W  +  g0 * b^T  +  (alpha*g_e (.) (x@A_e)) @ Bcat
// where Bcat = lora_B viewed as a contiguous 128 x 4096 matrix.
// Compute path: bf16x3 split (hi*hi + hi*lo + lo*hi) on
// v_wmma_f32_16x16x32_bf16 for ~fp32 accuracy at bf16 matrix rate / 3.
// A-tile staging uses the Tensor Data Mover (tensor_load_to_lds) with
// double-buffered LDS so the DMA overlaps the WMMA stream.
// ---------------------------------------------------------------------------

#define N_ROWS 4096
#define DD     4096
#define OO     4096
#define EE     8
#define RR     16

typedef __attribute__((ext_vector_type(16))) __bf16 v16bf;
typedef __attribute__((ext_vector_type(8)))  float  v8f;
typedef unsigned int u32x4 __attribute__((ext_vector_type(4)));
typedef int          i32x4 __attribute__((ext_vector_type(4)));
typedef int          i32x8 __attribute__((ext_vector_type(8)));

static __device__ __forceinline__ unsigned short f2bf(float f) {
  // round-to-nearest-even fp32 -> bf16
  unsigned u = __float_as_uint(f);
  u += 0x7fffu + ((u >> 16) & 1u);
  return (unsigned short)(u >> 16);
}
static __device__ __forceinline__ float bf2f(unsigned short h) {
  return __uint_as_float(((unsigned)h) << 16);
}
static __device__ __forceinline__ v8f vzero() {
  v8f z;
#pragma unroll
  for (int i = 0; i < 8; ++i) z[i] = 0.f;
  return z;
}

union Frag {
  v16bf v;
  uint4 q[2];
};

// ---------------------------------------------------------------------------
// TDM: issue a 2D tensor_load_to_lds of a (tile_d1 rows x tile_d0 elems)
// bf16 tile from global (row stride = stride0 elements) into LDS at lds_off.
// D# bitfields per CDNA5 ISA:
//   group0: count=1 | lds_addr | global_addr | type=2
//   group1: data_size=1 (2B), tensor_dim0/1, tile_dim0/1, tensor_dim0_stride
// Generic LDS pointers carry the LDS byte offset in their low 32 bits
// (ISA §10.2 aperture rule), which is what D#.lds_addr wants.
// ---------------------------------------------------------------------------
static __device__ __forceinline__ void tdm_load_bf16_2d(
    const unsigned short* gsrc, unsigned lds_off,
    unsigned tensor_d0, unsigned tensor_d1,
    unsigned tile_d0, unsigned tile_d1,
    unsigned long long stride0 /* elements */) {
  unsigned long long ga = (unsigned long long)(const void*)gsrc;
  u32x4 g0;
  g0[0] = 1u;                                   // count=1 (valid user D#)
  g0[1] = lds_off;                              // lds_addr (bytes)
  g0[2] = (unsigned)(ga & 0xffffffffu);         // global_addr[31:0]
  g0[3] = (unsigned)((ga >> 32) & 0x01ffffffu)  // global_addr[56:32]
          | (2u << 30);                         // type=2 ("image")
  i32x8 g1;
  g1[0] = (int)(1u << 16);                      // workgroup_mask=0, data_size=1 (2B)
  g1[1] = (int)((tensor_d0 & 0xffffu) << 16);   // [63:48] = tensor_dim0 lo16
  g1[2] = (int)((tensor_d0 >> 16) | ((tensor_d1 & 0xffffu) << 16));
  g1[3] = (int)((tensor_d1 >> 16) | (tile_d0 << 16));  // [127:112] tile_dim0
  g1[4] = (int)(tile_d1 & 0xffffu);             // tile_dim1 (tile_dim2=0)
  g1[5] = (int)(stride0 & 0xffffffffull);       // tensor_dim0_stride lo32
  g1[6] = (int)((stride0 >> 32) & 0xffffull);   // stride hi16; dim1_stride=0
  g1[7] = 0;
  i32x4 z4 = {0, 0, 0, 0};
#if __has_include(<hip/amd_detail/amd_gfx1250_TDM.h>)
  i32x8 z8 = {0, 0, 0, 0, 0, 0, 0, 0};
  __builtin_amdgcn_tensor_load_to_lds(g0, g1, z4, z4, z8, 0);
#else
  __builtin_amdgcn_tensor_load_to_lds(g0, g1, z4, z4, 0);
#endif
}

// ---------------------------------------------------------------------------
// Kernel 1: gating. One block per row: 9 dots of length D, top-2 over the 8
// LoRA logits, softmax over [base_logit, sparse_lora_logits] (non-top -> 0,
// NOT -inf, per reference). Writes gates[n][0..8].
// ---------------------------------------------------------------------------
__global__ void gate_kernel(const float* __restrict__ x,
                            const float* __restrict__ gw,
                            float* __restrict__ gates) {
  const int n = blockIdx.x;
  const int tid = threadIdx.x;
  float acc[9];
#pragma unroll
  for (int j = 0; j < 9; ++j) acc[j] = 0.f;
  const float* xr = x + (size_t)n * DD;
  for (int d = tid; d < DD; d += 256) {
    float xv = xr[d];
#pragma unroll
    for (int j = 0; j < 9; ++j) acc[j] += xv * gw[(size_t)j * DD + d];
  }
  __shared__ float red[256];
  __shared__ float logits[9];
  for (int j = 0; j < 9; ++j) {
    red[tid] = acc[j];
    __syncthreads();
    for (int s = 128; s > 0; s >>= 1) {
      if (tid < s) red[tid] += red[tid + s];
      __syncthreads();
    }
    if (tid == 0) logits[j] = red[0];
    __syncthreads();
  }
  if (tid == 0) {
    int i1 = -1, i2 = -1;
    float v1 = -INFINITY, v2 = -INFINITY;
    for (int e = 0; e < EE; ++e) {
      float v = logits[1 + e];
      if (v > v1) { v2 = v1; i2 = i1; v1 = v; i1 = e; }
      else if (v > v2) { v2 = v; i2 = e; }
    }
    float z[9];
    z[0] = logits[0];
    for (int e = 0; e < EE; ++e)
      z[1 + e] = (e == i1 || e == i2) ? logits[1 + e] : 0.f;
    float mx = z[0];
    for (int j = 1; j < 9; ++j) mx = fmaxf(mx, z[j]);
    float s = 0.f;
    for (int j = 0; j < 9; ++j) { z[j] = __expf(z[j] - mx); s += z[j]; }
    float inv = 1.f / s;
    for (int j = 0; j < 9; ++j) gates[(size_t)n * 9 + j] = z[j] * inv;
  }
}

// ---------------------------------------------------------------------------
// Kernel 2: x' = g0 (.) x, split to bf16 hi/lo. 4 elements / thread.
// ---------------------------------------------------------------------------
__global__ void scale_split_x_kernel(const float* __restrict__ x,
                                     const float* __restrict__ gates,
                                     unsigned short* __restrict__ xh,
                                     unsigned short* __restrict__ xl) {
  size_t base = ((size_t)blockIdx.x * 256 + threadIdx.x) * 4;
  int row = (int)(base >> 12);  // D = 4096
  float g = gates[(size_t)row * 9];
  float4 v = *(const float4*)(x + base);
  float a[4] = {v.x, v.y, v.z, v.w};
#pragma unroll
  for (int i = 0; i < 4; ++i) {
    float s = a[i] * g;
    unsigned short h = f2bf(s);
    xh[base + i] = h;
    xl[base + i] = f2bf(s - bf2f(h));
  }
}

// ---------------------------------------------------------------------------
// Kernel 3: t'[n][e*16+r] = alpha * g_e[n] * (x @ A_e)[n][r], bf16x3 WMMA.
// One block = 128 rows x all 8 experts; wave e owns expert e (exactly one
// 16-wide WMMA tile since R=16). Output split to bf16 hi/lo for kernel 4.
// ---------------------------------------------------------------------------
__global__ void lora_t_kernel(const float* __restrict__ x,
                              const float* __restrict__ loraA,
                              const float* __restrict__ gates,
                              unsigned short* __restrict__ th,
                              unsigned short* __restrict__ tl) {
  __shared__ unsigned short Xh[128][32], Xl[128][32];
  __shared__ unsigned short Bh[EE][16][32], Bl[EE][16][32];
  const int tid = threadIdx.x;
  const int wave = tid >> 5, lane = tid & 31;
  const int lrow = lane & 15, khalf = (lane >> 4) << 3;
  const int r0 = blockIdx.x * 128;

  v8f acc[8];
#pragma unroll
  for (int m = 0; m < 8; ++m) acc[m] = vzero();

  for (int kc = 0; kc < DD / 32; ++kc) {
    const int k0 = kc * 32;
    {  // stage X tile (128x32 f32 -> bf16 hi/lo)
      int row = tid >> 1, col = (tid & 1) * 16;
      const float* p = x + (size_t)(r0 + row) * DD + k0 + col;
#pragma unroll
      for (int i = 0; i < 16; ++i) {
        float v = p[i];
        unsigned short h = f2bf(v);
        Xh[row][col + i] = h;
        Xl[row][col + i] = f2bf(v - bf2f(h));
      }
    }
    {  // stage all 8 expert A tiles (32x16 each), N-major in LDS
      int e = tid >> 5, kr = tid & 31;
      const float* p = loraA + ((size_t)e * DD + (size_t)(k0 + kr)) * RR;
#pragma unroll
      for (int r = 0; r < RR; ++r) {
        float v = p[r];
        unsigned short h = f2bf(v);
        Bh[e][r][kr] = h;
        Bl[e][r][kr] = f2bf(v - bf2f(h));
      }
    }
    __syncthreads();

    Frag bh, bl;
    bh.q[0] = *(const uint4*)&Bh[wave][lrow][khalf];
    bh.q[1] = *(const uint4*)&Bh[wave][lrow][khalf + 16];
    bl.q[0] = *(const uint4*)&Bl[wave][lrow][khalf];
    bl.q[1] = *(const uint4*)&Bl[wave][lrow][khalf + 16];
#pragma unroll
    for (int m = 0; m < 8; ++m) {
      Frag ah, al;
      ah.q[0] = *(const uint4*)&Xh[m * 16 + lrow][khalf];
      ah.q[1] = *(const uint4*)&Xh[m * 16 + lrow][khalf + 16];
      al.q[0] = *(const uint4*)&Xl[m * 16 + lrow][khalf];
      al.q[1] = *(const uint4*)&Xl[m * 16 + lrow][khalf + 16];
      acc[m] = __builtin_amdgcn_wmma_f32_16x16x32_bf16(false, ah.v, false, bh.v,
                                                       (short)0, acc[m], false, false);
      acc[m] = __builtin_amdgcn_wmma_f32_16x16x32_bf16(false, ah.v, false, bl.v,
                                                       (short)0, acc[m], false, false);
      acc[m] = __builtin_amdgcn_wmma_f32_16x16x32_bf16(false, al.v, false, bh.v,
                                                       (short)0, acc[m], false, false);
    }
    __syncthreads();
  }

  // epilogue: scale by alpha*g_e and split to bf16 hi/lo
  const int e = wave;
  const int cr = lane & 15;
  const int rofs = (lane >> 4) << 3;
#pragma unroll
  for (int m = 0; m < 8; ++m) {
#pragma unroll
    for (int i = 0; i < 8; ++i) {
      int row = r0 + m * 16 + rofs + i;
      float g = gates[(size_t)row * 9 + 1 + e];  // ALPHA == 1
      float v = acc[m][i] * g;
      unsigned short h = f2bf(v);
      size_t o = (size_t)row * 128 + (size_t)e * 16 + cr;
      th[o] = h;
      tl[o] = f2bf(v - bf2f(h));
    }
  }
}

// ---------------------------------------------------------------------------
// Kernel 4: fused GEMM. out[128x128 tile] = x'split @ W + t'split @ Bcat
//           + g0 * bias.  8 waves; wave = 64x32 sub-tile (4x2 WMMA tiles).
// K loop: 128 chunks over D (A = xh/xl, B = W split on the fly) then 4
// chunks over E*R=128 (A = th/tl, B = lora_B split on the fly).
// A tiles: TDM tensor_load_to_lds, double-buffered, issued by wave 0 and
// pipelined one chunk ahead (TENSORcnt completes in-order -> wait <= 2).
// B tiles: VALU f32->bf16 hi/lo split during staging + global_prefetch_b8.
// ---------------------------------------------------------------------------
__global__ void main_gemm_kernel(const unsigned short* __restrict__ xh,
                                 const unsigned short* __restrict__ xl,
                                 const unsigned short* __restrict__ th,
                                 const unsigned short* __restrict__ tl,
                                 const float* __restrict__ W,
                                 const float* __restrict__ loraB,
                                 const float* __restrict__ bias,
                                 const float* __restrict__ gates,
                                 float* __restrict__ out) {
  __shared__ unsigned short Ah[2][128][32], Al[2][128][32];  // double-buffered
  __shared__ unsigned short Bh[128][32], Bl[128][32];        // N-major
  const int tid = threadIdx.x;
  const int wave = tid >> 5, lane = tid & 31;
  const int waveM = wave >> 2, waveN = wave & 3;
  const int lrow = lane & 15, khalf = (lane >> 4) << 3;
  const int c0 = blockIdx.x * 128, r0 = blockIdx.y * 128;

  v8f acc[4][2];
#pragma unroll
  for (int m = 0; m < 4; ++m)
#pragma unroll
    for (int n = 0; n < 2; ++n) acc[m][n] = vzero();

  const int KC0 = DD / 32;         // 128 chunks over D
  const int KCT = KC0 + 128 / 32;  // + 4 chunks over E*R

  // Prologue: wave 0 kicks off the TDM loads for chunk 0 into buffer 0.
  if (wave == 0) {
    tdm_load_bf16_2d(xh + (size_t)r0 * DD,
                     (unsigned)(uintptr_t)&Ah[0][0][0], DD, 128, 32, 128, DD);
    tdm_load_bf16_2d(xl + (size_t)r0 * DD,
                     (unsigned)(uintptr_t)&Al[0][0][0], DD, 128, 32, 128, DD);
  }

  for (int kc = 0; kc < KCT; ++kc) {
    const bool seg0 = kc < KC0;
    const int k0 = seg0 ? kc * 32 : (kc - KC0) * 32;
    const float* Bsrc = seg0 ? W : loraB;
    const int cur = kc & 1;

    {  // stage B tile: f32 -> bf16 hi/lo, store N-major
      int kr = tid >> 3, cb = (tid & 7) * 16;
      const float* p = Bsrc + (size_t)(k0 + kr) * OO + c0 + cb;
#pragma unroll
      for (int i = 0; i < 16; ++i) {
        float v = p[i];
        unsigned short h = f2bf(v);
        Bh[cb + i][kr] = h;
        Bl[cb + i][kr] = f2bf(v - bf2f(h));
      }
    }
    if (kc + 1 < KCT) {  // prefetch next B tile -> global_prefetch_b8
      bool ns0 = (kc + 1) < KC0;
      int nk0 = ns0 ? (kc + 1) * 32 : (kc + 1 - KC0) * 32;
      const float* np = (ns0 ? W : loraB) +
                        (size_t)(nk0 + (tid >> 3)) * OO + c0 + (tid & 7) * 16;
      __builtin_prefetch(np, 0, 0);
    }

    if (wave == 0) {
      if (kc + 1 < KCT) {  // TDM for next chunk into the other buffer
        bool ns0 = (kc + 1) < KC0;
        int nk0 = ns0 ? (kc + 1) * 32 : (kc + 1 - KC0) * 32;
        const unsigned short* nah = (ns0 ? xh : th);
        const unsigned short* nal = (ns0 ? xl : tl);
        unsigned long long nlda = ns0 ? DD : 128;
        int nxt = cur ^ 1;
        tdm_load_bf16_2d(nah + (size_t)r0 * nlda + nk0,
                         (unsigned)(uintptr_t)&Ah[nxt][0][0],
                         (unsigned)nlda, 128, 32, 128, nlda);
        tdm_load_bf16_2d(nal + (size_t)r0 * nlda + nk0,
                         (unsigned)(uintptr_t)&Al[nxt][0][0],
                         (unsigned)nlda, 128, 32, 128, nlda);
        // current buffer's 2 loads are the oldest; in-order completion
        // means "<= 2 outstanding" guarantees they are done.
        __builtin_amdgcn_s_wait_tensorcnt(2);
      } else {
        __builtin_amdgcn_s_wait_tensorcnt(0);
      }
    }
    __syncthreads();  // DMA'd A tile + staged B tile visible to all waves

    Frag bhf[2], blf[2];
#pragma unroll
    for (int n = 0; n < 2; ++n) {
      int cc = waveN * 32 + n * 16 + lrow;
      bhf[n].q[0] = *(const uint4*)&Bh[cc][khalf];
      bhf[n].q[1] = *(const uint4*)&Bh[cc][khalf + 16];
      blf[n].q[0] = *(const uint4*)&Bl[cc][khalf];
      blf[n].q[1] = *(const uint4*)&Bl[cc][khalf + 16];
    }
#pragma unroll
    for (int m = 0; m < 4; ++m) {
      int rr = waveM * 64 + m * 16 + lrow;
      Frag ahf, alf;
      ahf.q[0] = *(const uint4*)&Ah[cur][rr][khalf];
      ahf.q[1] = *(const uint4*)&Ah[cur][rr][khalf + 16];
      alf.q[0] = *(const uint4*)&Al[cur][rr][khalf];
      alf.q[1] = *(const uint4*)&Al[cur][rr][khalf + 16];
#pragma unroll
      for (int n = 0; n < 2; ++n) {
        acc[m][n] = __builtin_amdgcn_wmma_f32_16x16x32_bf16(
            false, ahf.v, false, bhf[n].v, (short)0, acc[m][n], false, false);
        acc[m][n] = __builtin_amdgcn_wmma_f32_16x16x32_bf16(
            false, ahf.v, false, blf[n].v, (short)0, acc[m][n], false, false);
        acc[m][n] = __builtin_amdgcn_wmma_f32_16x16x32_bf16(
            false, alf.v, false, bhf[n].v, (short)0, acc[m][n], false, false);
      }
    }
    __syncthreads();
  }

  // epilogue: + g0 * bias, write fp32 out
  const int rofs = (lane >> 4) << 3;
#pragma unroll
  for (int m = 0; m < 4; ++m) {
#pragma unroll
    for (int n = 0; n < 2; ++n) {
      int col = c0 + waveN * 32 + n * 16 + (lane & 15);
      float bv = bias[col];
#pragma unroll
      for (int i = 0; i < 8; ++i) {
        int row = r0 + waveM * 64 + m * 16 + rofs + i;
        float g0 = gates[(size_t)row * 9];
        out[(size_t)row * OO + col] = acc[m][n][i] + g0 * bv;
      }
    }
  }
}

// ---------------------------------------------------------------------------
extern "C" void kernel_launch(void* const* d_in, const int* in_sizes, int n_in,
                              void* d_out, int out_size, void* d_ws, size_t ws_size,
                              hipStream_t stream) {
  const float* x     = (const float*)d_in[0];  // (N, D)
  const float* W     = (const float*)d_in[1];  // (D, O)
  const float* bias  = (const float*)d_in[2];  // (O,)
  const float* loraA = (const float*)d_in[3];  // (E, D, R)
  const float* loraB = (const float*)d_in[4];  // (E, R, O) == (E*R, O) contiguous
  const float* gw    = (const float*)d_in[5];  // (E+1, D)
  // d_in[6] = bvv — unused by the reference computation.
  float* out = (float*)d_out;

  char* ws = (char*)d_ws;
  float* gates = (float*)ws;  // N*9 f32
  size_t off = (size_t)N_ROWS * 9 * sizeof(float);
  unsigned short* xh = (unsigned short*)(ws + off); off += (size_t)N_ROWS * DD * 2;
  unsigned short* xl = (unsigned short*)(ws + off); off += (size_t)N_ROWS * DD * 2;
  unsigned short* th = (unsigned short*)(ws + off); off += (size_t)N_ROWS * 128 * 2;
  unsigned short* tl = (unsigned short*)(ws + off);

  gate_kernel<<<N_ROWS, 256, 0, stream>>>(x, gw, gates);
  scale_split_x_kernel<<<(N_ROWS * DD) / 1024, 256, 0, stream>>>(x, gates, xh, xl);
  lora_t_kernel<<<N_ROWS / 128, 256, 0, stream>>>(x, loraA, gates, th, tl);
  dim3 grid(OO / 128, N_ROWS / 128);
  main_gemm_kernel<<<grid, 256, 0, stream>>>(xh, xl, th, tl, W, loraB, bias, gates, out);
}
